// TorsionNet_83786222011180
// MI455X (gfx1250) — matrix operand order repaired and checked by reading, not compiled, not run on hardware
//
#include <hip/hip_runtime.h>
#include <math.h>

typedef __bf16 bf16;
typedef __attribute__((ext_vector_type(16))) __bf16 v16bf;
typedef __attribute__((ext_vector_type(8)))  __bf16 v8bf;
typedef __attribute__((ext_vector_type(8)))  float  v8f;

static constexpr int NNODE = 30000;
static constexpr int NEDGE = 400000;
static constexpr int NTOR  = 50000;
static constexpr int NTW   = 200000;
static constexpr float PI_F = 3.14159265358979323846f;

#define DEV __device__ __forceinline__

// ---------------- WMMA fragment loaders (ISA 7.12.2 layouts) ----------------
// A (16x32 bf16): lanes 0-15 row M=lane hold K[0..8)+K[16..24); lanes 16-31 hold K[8..16)+K[24..32)
DEV v16bf frag_a(const bf16* p, int half) {
  v8bf lo = *(const v8bf*)(p + half * 8);
  v8bf hi = *(const v8bf*)(p + 16 + half * 8);
  return __builtin_shufflevector(lo, hi, 0,1,2,3,4,5,6,7,8,9,10,11,12,13,14,15);
}
// B (32x16 bf16): lane n<16 holds col n K[0..8)+K[16..24); lane n+16 holds col n K[8..16)+K[24..32)
DEV v16bf frag_b(const bf16* p, int half) {
  const bf16* q = p + half * 8;
  v8bf lo = *(const v8bf*)(q);
  v8bf hi = *(const v8bf*)(q + 16);
  return __builtin_shufflevector(lo, hi, 0,1,2,3,4,5,6,7,8,9,10,11,12,13,14,15);
}

// ---------------- small utility kernels ----------------
__global__ void zero_f32(float* __restrict__ p, int n) {
  int i = blockIdx.x * blockDim.x + threadIdx.x;
  if (i < n) p[i] = 0.0f;
}
__global__ void cast_bf16(const float* __restrict__ s, bf16* __restrict__ d, int n) {
  int i = blockIdx.x * blockDim.x + threadIdx.x;
  if (i < n) d[i] = (bf16)s[i];
}
__global__ void wt_transpose(const float* __restrict__ W, int K, int N,
                             bf16* __restrict__ Wt, int Kpad) {
  int i = blockIdx.x * blockDim.x + threadIdx.x;
  if (i >= N * Kpad) return;
  int n = i / Kpad, k = i % Kpad;
  Wt[i] = (k < K) ? (bf16)W[(size_t)k * N + n] : (bf16)0.0f;
}
// gate W1 (193x64): reorder so k<128 -> node rows 64..191, k in [128,192) -> edge rows 0..63
__global__ void wt_gt(const float* __restrict__ W, bf16* __restrict__ Wt) {
  int i = blockIdx.x * blockDim.x + threadIdx.x;
  if (i >= 64 * 192) return;
  int n = i / 192, k = i % 192;
  int src = (k < 128) ? (64 + k) : (k - 128);
  Wt[i] = (bf16)W[src * 64 + n];
}

// ---------------- stage A: per-torsion precompute ----------------
__global__ void torsion_pre(const int* __restrict__ eidx, const int* __restrict__ tanno,
                            const float* __restrict__ pos, int* __restrict__ torLeft,
                            float* __restrict__ unit) {
  int t = blockIdx.x * blockDim.x + threadIdx.x;
  if (t >= NTOR) return;
  int e = tanno[NTOR + t];
  int l = eidx[e], r = eidx[NEDGE + e];
  float dx = pos[3*l]   - pos[3*r];
  float dy = pos[3*l+1] - pos[3*r+1];
  float dz = pos[3*l+2] - pos[3*r+2];
  float inv = 1.0f / (sqrtf(dx*dx + dy*dy + dz*dz) + 1e-6f);
  torLeft[t] = l;
  unit[3*t] = dx*inv; unit[3*t+1] = dy*inv; unit[3*t+2] = dz*inv;
}

// ---------------- stage B: build 200000x480 bf16 torque features ----------------
__global__ __launch_bounds__(256)
void build_xtq(const float* __restrict__ h_node, const float* __restrict__ h_edge,
               const float* __restrict__ pos, const float* __restrict__ force,
               const int* __restrict__ eidx, const int* __restrict__ tanno,
               const int* __restrict__ twanno,
               const int* __restrict__ torLeft, const float* __restrict__ unit,
               bf16* __restrict__ X, float* __restrict__ tqRaw, float* __restrict__ vecTe,
               int* __restrict__ twNode, int* __restrict__ torEndA) {
  const int lane = threadIdx.x & 31;
  const int wave = threadIdx.x >> 5;
  const int t = blockIdx.x * 8 + wave;
  if (t >= NTW) return;
  const int itw = twanno[t];
  const int te  = twanno[NTW + t];
  const int nI  = eidx[te];
  const int eI  = eidx[NEDGE + te];
  const float ux = unit[3*itw], uy = unit[3*itw+1], uz = unit[3*itw+2];
  const float vx = pos[3*nI]   - pos[3*eI];
  const float vy = pos[3*nI+1] - pos[3*eI+1];
  const float vz = pos[3*nI+2] - pos[3*eI+2];
  const float fx = force[3*nI], fy = force[3*nI+1], fz = force[3*nI+2];
  const float du = vx*ux + vy*uy + vz*uz;
  const float rx = vx - du*ux, ry = vy - du*uy, rz = vz - du*uz;
  const float rad = sqrtf(rx*rx + ry*ry + rz*rz);
  const float fu = fx*ux + fy*uy + fz*uz;
  const float gx = fx - fu*ux, gy = fy - fu*uy, gz = fz - fu*uz;   // force_tangent
  const float qx = ry*gz - rz*gy, qy = rz*gx - rx*gz, qz = rx*gy - ry*gx; // torque
  bf16* row = X + (size_t)t * 480;
  const float* hs = h_node + (size_t)nI * 128;
  for (int c = lane; c < 128; c += 32) row[c] = (bf16)hs[c];
  const float* es = h_edge + (size_t)te * 64;
  for (int c = lane; c < 64; c += 32) row[128 + c] = (bf16)es[c];
  const int tl = torLeft[itw];
  const float* ls = h_node + (size_t)tl * 128;
  for (int c = lane; c < 128; c += 32) row[192 + c] = (bf16)ls[c];
  const int toe = tanno[NTOR + itw];
  const float* os = h_edge + (size_t)toe * 64;
  for (int c = lane; c < 64; c += 32) row[320 + c] = (bf16)os[c];
  const float step = 10.0f / 63.0f;
  const float coeff = -0.5f / (step * step);
  for (int j = lane; j < 64; j += 32) {
    float d = rad - step * (float)j;
    row[384 + j] = (bf16)__expf(coeff * d * d);
  }
  if (lane == 0) {
    row[448] = (bf16)sqrtf(fx*fx + fy*fy + fz*fz);
    row[449] = (bf16)sqrtf(gx*gx + gy*gy + gz*gz);
    row[450] = (bf16)sqrtf(qx*qx + qy*qy + qz*qz);
    tqRaw[3*t] = qx; tqRaw[3*t+1] = qy; tqRaw[3*t+2] = qz;
    vecTe[3*t] = vx; vecTe[3*t+1] = vy; vecTe[3*t+2] = vz;
    twNode[t] = nI; torEndA[t] = eI;
  }
  for (int c = 451 + lane; c < 480; c += 32) row[c] = (bf16)0.0f;
}

// build 50000x160 bf16 angle features: [len_torque, h_node_new[tor_left], pad]
__global__ __launch_bounds__(256)
void build_xan(const float* __restrict__ lenT, const float* __restrict__ hnew,
               const int* __restrict__ torLeft, bf16* __restrict__ X) {
  const int lane = threadIdx.x & 31;
  const int wave = threadIdx.x >> 5;
  const int t = blockIdx.x * 8 + wave;
  if (t >= NTOR) return;
  bf16* row = X + (size_t)t * 160;
  if (lane == 0) row[0] = (bf16)lenT[t];
  const float* src = hnew + (size_t)torLeft[t] * 128;
  for (int c = lane; c < 128; c += 32) row[1 + c] = (bf16)src[c];
  for (int c = 129 + lane; c < 160; c += 32) row[c] = (bf16)0.0f;
}

// ---------------- fused MLP2 kernel: GEMM1 + bias + LN + ReLU (+ GEMM2) ----------------
static constexpr int EPI_TQ = 0, EPI_AN = 1, EPI_NN = 2, EPI_PROD = 3, EPI_SIG = 4, EPI_HNODE = 5;

template<int K1, int H, int N2, int EPI>
__global__ __launch_bounds__(256)
void mlp2_kernel(const bf16* __restrict__ X, int lda, int M,
                 const bf16* __restrict__ W1t,                       // [H][K1] bf16
                 const float* __restrict__ b1, const float* __restrict__ g1,
                 const float* __restrict__ be1,
                 const bf16* __restrict__ W2t,                       // [N2][H] bf16 (N2>=16)
                 const float* __restrict__ w2v,                      // [H] (N2==1)
                 const float* __restrict__ b2,
                 float* __restrict__ outF, bf16* __restrict__ outB, int ldo,
                 const int* __restrict__ idx, const bf16* __restrict__ auxB,
                 const float* __restrict__ auxF, int auxLd) {
  constexpr int NT1 = H / 16;
  const int lane = threadIdx.x & 31;
  const int wave = threadIdx.x >> 5;
  const int half = lane >> 4;
  const int ml   = lane & 15;
  const int tile = blockIdx.x * (blockDim.x >> 5) + wave;
  const int r0 = tile * 16;
  if (r0 >= M) return;

  int arow = r0 + ml; if (arow >= M) arow = M - 1;
  const bf16* aptr = X + (size_t)arow * lda;
  const bf16* bcol = W1t + (size_t)ml * K1;

  v8f acc[NT1];
#pragma unroll
  for (int n = 0; n < NT1; ++n)
#pragma unroll
    for (int j = 0; j < 8; ++j) acc[n][j] = 0.0f;

  for (int k0 = 0; k0 < K1; k0 += 32) {
    v16bf a = frag_a(aptr + k0, half);
#pragma unroll
    for (int n = 0; n < NT1; ++n) {
      v16bf b = frag_b(bcol + (size_t)(n * 16) * K1 + k0, half);
      acc[n] = __builtin_amdgcn_wmma_f32_16x16x32_bf16(
          false, a, false, b, (short)0, acc[n], false, false);
    }
  }

  // bias (+ optional pre-LN aggr add), LayerNorm over H, ReLU
#pragma unroll
  for (int n = 0; n < NT1; ++n) {
    const float bb = b1[n * 16 + ml];
#pragma unroll
    for (int m = 0; m < 8; ++m) acc[n][m] += bb;
  }
  if constexpr (EPI == EPI_HNODE) {
#pragma unroll
    for (int n = 0; n < NT1; ++n) {
      const int col = n * 16 + ml;
#pragma unroll
      for (int m = 0; m < 8; ++m) {
        int row = r0 + m + half * 8; if (row >= M) row = M - 1;
        acc[n][m] += auxF[(size_t)row * auxLd + col];
      }
    }
  }
  float s[8], ss[8];
#pragma unroll
  for (int m = 0; m < 8; ++m) { s[m] = 0.0f; ss[m] = 0.0f; }
#pragma unroll
  for (int n = 0; n < NT1; ++n)
#pragma unroll
    for (int m = 0; m < 8; ++m) { float x = acc[n][m]; s[m] += x; ss[m] += x * x; }
  for (int off = 1; off < 16; off <<= 1) {
#pragma unroll
    for (int m = 0; m < 8; ++m) {
      s[m]  += __shfl_xor(s[m],  off, 32);
      ss[m] += __shfl_xor(ss[m], off, 32);
    }
  }
#pragma unroll
  for (int m = 0; m < 8; ++m) {
    const float mean = s[m] * (1.0f / (float)H);
    const float var  = ss[m] * (1.0f / (float)H) - mean * mean;
    s[m] = mean; ss[m] = rsqrtf(var + 1e-5f);
  }
#pragma unroll
  for (int n = 0; n < NT1; ++n) {
    const float gg = g1[n * 16 + ml], bb = be1[n * 16 + ml];
#pragma unroll
    for (int m = 0; m < 8; ++m) {
      float x = (acc[n][m] - s[m]) * ss[m] * gg + bb;
      acc[n][m] = x > 0.0f ? x : 0.0f;
    }
  }

  if constexpr (N2 == 1) {
    // second layer is a dot product (H -> 1): reduce in-register
    float d[8];
#pragma unroll
    for (int m = 0; m < 8; ++m) d[m] = 0.0f;
#pragma unroll
    for (int n = 0; n < NT1; ++n) {
      const float w = w2v[n * 16 + ml];
#pragma unroll
      for (int m = 0; m < 8; ++m) d[m] += acc[n][m] * w;
    }
    for (int off = 1; off < 16; off <<= 1) {
#pragma unroll
      for (int m = 0; m < 8; ++m) d[m] += __shfl_xor(d[m], off, 32);
    }
    if (ml == 0) {
      const float bb = b2[0];
#pragma unroll
      for (int m = 0; m < 8; ++m) {
        const int row = r0 + m + half * 8;
        if (row < M) {
          float v = d[m] + bb;
          if constexpr (EPI == EPI_AN) {
            const float dr = auxF[row];
            const float sg = dr > 0.0f ? 1.0f : (dr < 0.0f ? -1.0f : 0.0f);
            v = (1.0f / (1.0f + __expf(-v))) * PI_F * sg;
          }
          outF[row] = v;
        }
      }
    }
  } else {
    // round-trip the ReLU'd tile through LDS to re-shape C-layout -> A-layout
    __shared__ __align__(16) bf16 lds[8 * 16 * H];
    bf16* myl = lds + wave * (16 * H);
#pragma unroll
    for (int n = 0; n < NT1; ++n) {
      const int col = n * 16 + ml;
#pragma unroll
      for (int m = 0; m < 8; ++m)
        myl[(m + half * 8) * H + col] = (bf16)acc[n][m];
    }
    asm volatile("s_wait_dscnt 0" ::: "memory");   // wave-local DS ordering (CDNA5 split counter)

    constexpr int NT2 = N2 / 16;
    v8f acc2[NT2];
#pragma unroll
    for (int n = 0; n < NT2; ++n)
#pragma unroll
      for (int j = 0; j < 8; ++j) acc2[n][j] = 0.0f;
    const bf16* b2col = W2t + (size_t)ml * H;
#pragma unroll
    for (int k0 = 0; k0 < H; k0 += 32) {
      v16bf a2 = frag_a(myl + ml * H + k0, half);
#pragma unroll
      for (int n = 0; n < NT2; ++n) {
        v16bf bfr = frag_b(b2col + (size_t)(n * 16) * H + k0, half);
        acc2[n] = __builtin_amdgcn_wmma_f32_16x16x32_bf16(
            false, a2, false, bfr, (short)0, acc2[n], false, false);
      }
    }
#pragma unroll
    for (int n = 0; n < NT2; ++n) {
      const int col = n * 16 + ml;
      const float bb = b2[col];
#pragma unroll
      for (int m = 0; m < 8; ++m) {
        const int row = r0 + m + half * 8;
        if (row < M) {
          float v = acc2[n][m] + bb;
          if constexpr (EPI == EPI_NN) {
            outB[(size_t)row * ldo + col] = (bf16)v;
          } else if constexpr (EPI == EPI_PROD) {
            const float hv = (float)auxB[(size_t)idx[row] * 64 + col];
            outB[(size_t)row * ldo + col] = (bf16)(v * hv);
          } else if constexpr (EPI == EPI_SIG) {
            outB[(size_t)row * ldo + col] = (bf16)(1.0f / (1.0f + __expf(-v)));
          } else { // EPI_HNODE stage-2: fp32 h_node_new
            outF[(size_t)row * ldo + col] = v;
          }
        }
      }
    }
  }
}

// ---------------- msg linear + gate + scatter (segment_sum into nodes) ----------------
__global__ __launch_bounds__(256)
void msg_scatter(const bf16* __restrict__ prod, const bf16* __restrict__ Wt,
                 const float* __restrict__ mb, const bf16* __restrict__ gate,
                 const int* __restrict__ torEndA, float* __restrict__ aggr, int M) {
  const int lane = threadIdx.x & 31;
  const int wave = threadIdx.x >> 5;
  const int half = lane >> 4;
  const int ml   = lane & 15;
  const int tile = blockIdx.x * (blockDim.x >> 5) + wave;
  const int r0 = tile * 16;
  if (r0 >= M) return;
  int arow = r0 + ml; if (arow >= M) arow = M - 1;
  const bf16* aptr = prod + (size_t)arow * 64;
  const bf16* bcol = Wt + (size_t)ml * 64;
  v8f acc[4];
#pragma unroll
  for (int n = 0; n < 4; ++n)
#pragma unroll
    for (int j = 0; j < 8; ++j) acc[n][j] = 0.0f;
#pragma unroll
  for (int k0 = 0; k0 < 64; k0 += 32) {
    v16bf a = frag_a(aptr + k0, half);
#pragma unroll
    for (int n = 0; n < 4; ++n) {
      v16bf b = frag_b(bcol + (size_t)(n * 16) * 64 + k0, half);
      acc[n] = __builtin_amdgcn_wmma_f32_16x16x32_bf16(
          false, a, false, b, (short)0, acc[n], false, false);
    }
  }
#pragma unroll
  for (int n = 0; n < 4; ++n) {
    const int col = n * 16 + ml;
    const float bb = mb[col];
#pragma unroll
    for (int m = 0; m < 8; ++m) {
      const int row = r0 + m + half * 8;
      if (row < M) {
        float v = (acc[n][m] + bb) * (float)gate[(size_t)row * 64 + col];
        atomicAdd(&aggr[(size_t)torEndA[row] * 64 + col], v);
      }
    }
  }
}

// ---------------- torque scatter / finalize, Rodrigues scatter / finalize ----------------
__global__ void torque_scatter(const float* __restrict__ scale, const float* __restrict__ tqRaw,
                               const int* __restrict__ itw, float* __restrict__ torAcc) {
  int t = blockIdx.x * blockDim.x + threadIdx.x;
  if (t >= NTW) return;
  float s = scale[t];
  int k = itw[t];
  atomicAdd(&torAcc[4*k+0], tqRaw[3*t+0] * s);
  atomicAdd(&torAcc[4*k+1], tqRaw[3*t+1] * s);
  atomicAdd(&torAcc[4*k+2], tqRaw[3*t+2] * s);
  atomicAdd(&torAcc[4*k+3], 1.0f);
}
__global__ void torsion_fin(const float* __restrict__ torAcc, const float* __restrict__ unit,
                            float* __restrict__ lenT, float* __restrict__ dir) {
  int t = blockIdx.x * blockDim.x + threadIdx.x;
  if (t >= NTOR) return;
  float c = torAcc[4*t+3]; c = c > 1.0f ? c : 1.0f;
  float x = torAcc[4*t] / c, y = torAcc[4*t+1] / c, z = torAcc[4*t+2] / c;
  lenT[t] = sqrtf(x*x + y*y + z*z);
  dir[t]  = x * unit[3*t] + y * unit[3*t+1] + z * unit[3*t+2];
}
__global__ void rod_scatter(const float* __restrict__ ang, const int* __restrict__ itw,
                            const float* __restrict__ unit, const float* __restrict__ vecTe,
                            const int* __restrict__ twNode, const int* __restrict__ torEndA,
                            const float* __restrict__ pos, float* __restrict__ posAcc) {
  int t = blockIdx.x * blockDim.x + threadIdx.x;
  if (t >= NTW) return;
  int k = itw[t];
  float a = ang[k], ca = cosf(a), sa = sinf(a), omc = 1.0f - ca;
  float ux = unit[3*k], uy = unit[3*k+1], uz = unit[3*k+2];
  float vx = vecTe[3*t], vy = vecTe[3*t+1], vz = vecTe[3*t+2];
  float cx = uy*vz - uz*vy, cy = uz*vx - ux*vz, cz = ux*vy - uy*vx;
  float ud = ux*vx + uy*vy + uz*vz;
  float rx = vx*ca + cx*sa + ux*ud*omc;
  float ry = vy*ca + cy*sa + uy*ud*omc;
  float rz = vz*ca + cz*sa + uz*ud*omc;
  int nI = twNode[t], eI = torEndA[t];
  float dx = pos[3*eI]   + rx - pos[3*nI];
  float dy = pos[3*eI+1] + ry - pos[3*nI+1];
  float dz = pos[3*eI+2] + rz - pos[3*nI+2];
  atomicAdd(&posAcc[4*nI+0], dx);
  atomicAdd(&posAcc[4*nI+1], dy);
  atomicAdd(&posAcc[4*nI+2], dz);
  atomicAdd(&posAcc[4*nI+3], 1.0f);
}
__global__ void pos_fin(const float* __restrict__ pos, const float* __restrict__ posAcc,
                        float* __restrict__ out) {
  int n = blockIdx.x * blockDim.x + threadIdx.x;
  if (n >= NNODE) return;
  float c = posAcc[4*n+3]; c = c > 1.0f ? c : 1.0f;
  out[3*n]   = pos[3*n]   + posAcc[4*n]   / c;
  out[3*n+1] = pos[3*n+1] + posAcc[4*n+1] / c;
  out[3*n+2] = pos[3*n+2] + posAcc[4*n+2] / c;
}

// ---------------- host launch ----------------
extern "C" void kernel_launch(void* const* d_in, const int* in_sizes, int n_in,
                              void* d_out, int out_size, void* d_ws, size_t ws_size,
                              hipStream_t stream) {
  (void)in_sizes; (void)n_in; (void)out_size; (void)ws_size;
  const float* h_node = (const float*)d_in[0];
  const float* pos    = (const float*)d_in[1];
  const float* force  = (const float*)d_in[2];
  const float* h_edge = (const float*)d_in[3];
  const int*   eidx   = (const int*)d_in[4];
  const int*   tanno  = (const int*)d_in[5];
  const int*   twanno = (const int*)d_in[6];
  const float *tq_W1=(const float*)d_in[7],  *tq_b1=(const float*)d_in[8],
              *tq_g =(const float*)d_in[9],  *tq_be=(const float*)d_in[10],
              *tq_W2=(const float*)d_in[11], *tq_b2=(const float*)d_in[12];
  const float *an_W1=(const float*)d_in[13], *an_b1=(const float*)d_in[14],
              *an_g =(const float*)d_in[15], *an_be=(const float*)d_in[16],
              *an_W2=(const float*)d_in[17], *an_b2=(const float*)d_in[18];
  const float *nn_W1=(const float*)d_in[19], *nn_b1=(const float*)d_in[20],
              *nn_g =(const float*)d_in[21], *nn_be=(const float*)d_in[22],
              *nn_W2=(const float*)d_in[23], *nn_b2=(const float*)d_in[24];
  const float *en_W1=(const float*)d_in[25], *en_b1=(const float*)d_in[26],
              *en_g =(const float*)d_in[27], *en_be=(const float*)d_in[28],
              *en_W2=(const float*)d_in[29], *en_b2=(const float*)d_in[30];
  const float *msg_W=(const float*)d_in[31], *msg_b=(const float*)d_in[32];
  const float *gt_W1=(const float*)d_in[33], *gt_b1=(const float*)d_in[34],
              *gt_g =(const float*)d_in[35], *gt_be=(const float*)d_in[36],
              *gt_W2=(const float*)d_in[37], *gt_b2=(const float*)d_in[38];
  const float *cl_W =(const float*)d_in[39], *cl_b =(const float*)d_in[40],
              *ln_g =(const float*)d_in[41], *ln_b =(const float*)d_in[42],
              *ot_W =(const float*)d_in[43], *ot_b =(const float*)d_in[44];

  char* p = (char*)d_ws;
  auto carve = [&](size_t bytes) -> void* {
    void* r = (void*)p;
    p += (bytes + 255) & ~(size_t)255;
    return r;
  };
  bf16* Xtq    = (bf16*)carve((size_t)NTW * 480 * 2);
  bf16* Xan    = (bf16*)carve((size_t)NTOR * 160 * 2);
  bf16* hnodeB = (bf16*)carve((size_t)NNODE * 128 * 2);
  bf16* hn     = (bf16*)carve((size_t)NNODE * 64 * 2);
  bf16* prod   = (bf16*)carve((size_t)NTW * 64 * 2);
  bf16* gate   = (bf16*)carve((size_t)NTW * 64 * 2);
  float* tqRaw  = (float*)carve((size_t)NTW * 3 * 4);
  float* vecTe  = (float*)carve((size_t)NTW * 3 * 4);
  float* scaleO = (float*)carve((size_t)NTW * 4);
  int*   twNode = (int*)carve((size_t)NTW * 4);
  int*   torEndA= (int*)carve((size_t)NTW * 4);
  int*   torLeft= (int*)carve((size_t)NTOR * 4);
  float* unit   = (float*)carve((size_t)NTOR * 3 * 4);
  float* torAcc = (float*)carve((size_t)NTOR * 4 * 4);
  float* lenT   = (float*)carve((size_t)NTOR * 4);
  float* dir    = (float*)carve((size_t)NTOR * 4);
  float* aggr   = (float*)carve((size_t)NNODE * 64 * 4);
  float* hnew   = (float*)carve((size_t)NNODE * 128 * 4);
  float* posAcc = (float*)carve((size_t)NNODE * 4 * 4);
  bf16* Wtq1 = (bf16*)carve(128 * 480 * 2);
  bf16* Wan1 = (bf16*)carve(64 * 160 * 2);
  bf16* Wnn1 = (bf16*)carve(64 * 128 * 2);
  bf16* Wnn2 = (bf16*)carve(64 * 64 * 2);
  bf16* Wen1 = (bf16*)carve(64 * 64 * 2);
  bf16* Wen2 = (bf16*)carve(64 * 64 * 2);
  bf16* Wmsg = (bf16*)carve(64 * 64 * 2);
  bf16* Wgt1 = (bf16*)carve(64 * 192 * 2);
  bf16* Wgt2 = (bf16*)carve(64 * 64 * 2);
  bf16* Wcl  = (bf16*)carve(64 * 128 * 2);
  bf16* Wot  = (bf16*)carve(128 * 64 * 2);

  auto T2 = [&](const float* W, int K, int N, bf16* Wt, int Kpad) {
    int tot = N * Kpad;
    wt_transpose<<<(tot + 255) / 256, 256, 0, stream>>>(W, K, N, Wt, Kpad);
  };
  T2(tq_W1, 451, 128, Wtq1, 480);
  T2(an_W1, 129, 64,  Wan1, 160);
  T2(nn_W1, 128, 64,  Wnn1, 128);
  T2(nn_W2, 64,  64,  Wnn2, 64);
  T2(en_W1, 64,  64,  Wen1, 64);
  T2(en_W2, 64,  64,  Wen2, 64);
  T2(msg_W, 64,  64,  Wmsg, 64);
  T2(gt_W2, 64,  64,  Wgt2, 64);
  T2(cl_W,  128, 64,  Wcl,  128);
  T2(ot_W,  64,  128, Wot,  64);
  wt_gt<<<(64 * 192 + 255) / 256, 256, 0, stream>>>(gt_W1, Wgt1);
  cast_bf16<<<(NNODE * 128 + 255) / 256, 256, 0, stream>>>(h_node, hnodeB, NNODE * 128);

  zero_f32<<<(NTOR * 4 + 255) / 256, 256, 0, stream>>>(torAcc, NTOR * 4);
  zero_f32<<<(NNODE * 64 + 255) / 256, 256, 0, stream>>>(aggr, NNODE * 64);
  zero_f32<<<(NNODE * 4 + 255) / 256, 256, 0, stream>>>(posAcc, NNODE * 4);

  torsion_pre<<<(NTOR + 255) / 256, 256, 0, stream>>>(eidx, tanno, pos, torLeft, unit);
  build_xtq<<<(NTW + 7) / 8, 256, 0, stream>>>(h_node, h_edge, pos, force, eidx, tanno, twanno,
                                               torLeft, unit, Xtq, tqRaw, vecTe, twNode, torEndA);

  auto mlpGrid = [](int M) { return ((M + 15) / 16 + 7) / 8; };

  // torque MLP: 451->128 (LN,ReLU) ->1  => per-row scale
  mlp2_kernel<480, 128, 1, EPI_TQ><<<mlpGrid(NTW), 256, 0, stream>>>(
      Xtq, 480, NTW, Wtq1, tq_b1, tq_g, tq_be, nullptr, tq_W2, tq_b2,
      scaleO, nullptr, 1, nullptr, nullptr, nullptr, 0);

  torque_scatter<<<(NTW + 255) / 256, 256, 0, stream>>>(scaleO, tqRaw, twanno, torAcc);
  torsion_fin<<<(NTOR + 255) / 256, 256, 0, stream>>>(torAcc, unit, lenT, dir);

  // node MLP: 128->64->64 (bf16 out)
  mlp2_kernel<128, 64, 64, EPI_NN><<<mlpGrid(NNODE), 256, 0, stream>>>(
      hnodeB, 128, NNODE, Wnn1, nn_b1, nn_g, nn_be, Wnn2, nullptr, nn_b2,
      nullptr, hn, 64, nullptr, nullptr, nullptr, 0);

  // edge MLP on gathered edges (view of Xtq cols 128:192), fused he * hn[twisted_node]
  mlp2_kernel<64, 64, 64, EPI_PROD><<<mlpGrid(NTW), 256, 0, stream>>>(
      Xtq + 128, 480, NTW, Wen1, en_b1, en_g, en_be, Wen2, nullptr, en_b2,
      nullptr, prod, 64, twNode, hn, nullptr, 0);

  // gate MLP on view of Xtq cols 0:192 with reordered weights, sigmoid stored
  mlp2_kernel<192, 64, 64, EPI_SIG><<<mlpGrid(NTW), 256, 0, stream>>>(
      Xtq, 480, NTW, Wgt1, gt_b1, gt_g, gt_be, Wgt2, nullptr, gt_b2,
      nullptr, gate, 64, nullptr, nullptr, nullptr, 0);

  msg_scatter<<<mlpGrid(NTW), 256, 0, stream>>>(prod, Wmsg, msg_b, gate, torEndA, aggr, NTW);

  // h_node_new = relu(LN(h_node@cl_W + cl_b + aggr)) @ ot_W + ot_b
  mlp2_kernel<128, 64, 128, EPI_HNODE><<<mlpGrid(NNODE), 256, 0, stream>>>(
      hnodeB, 128, NNODE, Wcl, cl_b, ln_g, ln_b, Wot, nullptr, ot_b,
      hnew, nullptr, 128, nullptr, nullptr, aggr, 64);

  build_xan<<<(NTOR + 7) / 8, 256, 0, stream>>>(lenT, hnew, torLeft, Xan);

  float* outPos = (float*)d_out;
  float* outAng = outPos + NNODE * 3;

  // angle MLP: 129->64->1, sigmoid * pi * sign(direction)
  mlp2_kernel<160, 64, 1, EPI_AN><<<mlpGrid(NTOR), 256, 0, stream>>>(
      Xan, 160, NTOR, Wan1, an_b1, an_g, an_be, nullptr, an_W2, an_b2,
      outAng, nullptr, 1, nullptr, nullptr, dir, 0);

  rod_scatter<<<(NTW + 255) / 256, 256, 0, stream>>>(outAng, twanno, unit, vecTe,
                                                     twNode, torEndA, pos, posAcc);
  pos_fin<<<(NNODE + 255) / 256, 256, 0, stream>>>(pos, posAcc, outPos);
}
